// QuantLeNet_33732673143601
// MI455X (gfx1250) — compile-verified, matrix operand order checked by reading, not run on hardware
//
#include <hip/hip_runtime.h>
#include <hip/hip_bf16.h>
#include <math.h>

// QuantLeNet (TextCNN) for MI455X / gfx1250.
// Convs lowered to GEMM on v_wmma_f32_16x16x32_f16. Activations are staged
// into LDS by the Tensor Data Mover (tensor_load_to_lds, double-buffered,
// TENSORcnt-synchronized), WMMA operands come from bank-conflict-free
// ds_load_b128 (TDM pad feature gives 80B row stride). Pooling is fused so
// conv activations never touch global memory.

typedef __attribute__((ext_vector_type(16))) _Float16 v16h;
typedef __attribute__((ext_vector_type(8)))  _Float16 v8h;
typedef __attribute__((ext_vector_type(8)))  float    v8f;
typedef __attribute__((ext_vector_type(4)))  unsigned int u32x4;
typedef __attribute__((ext_vector_type(8)))  int i32x8;
typedef __attribute__((ext_vector_type(4)))  int i32x4;

#define VV 50000
#define DD 300
#define DP 320      // D padded to multiple of 32 (WMMA K-chunk alignment)
#define FF 256
#define CC 2
#define BB 64
#define SS 512
#define SP 520      // S + 8 zero pad rows (tail tiles read t+j up to 515)
#define ROWB 80     // LDS row stride: 64B data + 16B TDM pad (conflict-free)
#define ABUF (SP * ROWB)        // one staging buffer: 41,600 B
#define CONV_SMEM (2 * ABUF)    // double buffer: 83,200 B

// ---- Tensor Data Mover: 2D tile load, 2-byte elements, optional LDS pad ----
// D# packing per CDNA5 ISA ch.8 (group0 128b, group1 256b; groups 2/3 zero
// for 2D tensors). lds_addr = low 32 bits of the generic LDS pointer.
__device__ __forceinline__ void tdm_load_2d_f16(unsigned int lds_addr, unsigned long long gaddr,
                                                unsigned int tensor_w, unsigned int tensor_h,
                                                unsigned int tile_w, unsigned int tile_h,
                                                unsigned long long row_stride,
                                                unsigned int pad_interval_code,
                                                unsigned int pad_amount_code,
                                                bool pad_en) {
  u32x4 g0;
  g0.x = 0x1u;                                            // count=1, user mode
  g0.y = lds_addr;                                        // LDS byte address
  g0.z = (unsigned int)gaddr;                             // global_addr[31:0]
  g0.w = ((unsigned int)(gaddr >> 32) & 0x01FFFFFFu)      // global_addr[56:32]
       | (2u << 30);                                      // type = 2 ("image")
  i32x8 g1;
  unsigned int w0 = (1u << 16);                           // data_size = 2 bytes
  if (pad_en) w0 |= (1u << 20) | (pad_interval_code << 22) | (pad_amount_code << 25);
  g1[0] = (int)w0;
  g1[1] = (int)((tensor_w & 0xFFFFu) << 16);              // tensor_dim0[15:0] @ bits 63:48
  g1[2] = (int)((tensor_w >> 16) | ((tensor_h & 0xFFFFu) << 16));  // dim0 hi | dim1 lo
  g1[3] = (int)((tensor_h >> 16) | (tile_w << 16));       // dim1 hi | tile_dim0
  g1[4] = (int)(tile_h & 0xFFFFu);                        // tile_dim1 (tile_dim2 = 0)
  g1[5] = (int)(unsigned int)(row_stride & 0xFFFFFFFFu);  // tensor_dim0_stride lo32
  g1[6] = (int)(unsigned int)((row_stride >> 32) & 0xFFFFu);
  g1[7] = 0;
  i32x4 gz = {0, 0, 0, 0};
#if __has_include(<hip/amd_detail/amd_gfx1250_TDM.h>)
  i32x8 gz8 = {0, 0, 0, 0, 0, 0, 0, 0};
  __builtin_amdgcn_tensor_load_to_lds(g0, g1, gz, gz, gz8, 0);
#else
  __builtin_amdgcn_tensor_load_to_lds(g0, g1, gz, gz, 0);
#endif
}

// ---------------- stage 1: per-tensor |w|max -> scale = max/127 ----------------
__global__ void scales_kernel(const float* __restrict__ w3, const float* __restrict__ w4,
                              const float* __restrict__ w5, const float* __restrict__ fcw,
                              float* __restrict__ scales) {
  __shared__ float red[256];
  const float* p; int n;
  switch (blockIdx.x) {
    case 0:  p = w3;  n = FF * DD * 3; break;
    case 1:  p = w4;  n = FF * DD * 4; break;
    case 2:  p = w5;  n = FF * DD * 5; break;
    default: p = fcw; n = CC * 3 * FF; break;
  }
  float m = 0.f;
  for (int i = threadIdx.x; i < n; i += 256) m = fmaxf(m, fabsf(p[i]));
  red[threadIdx.x] = m;
  __syncthreads();
  for (int s = 128; s > 0; s >>= 1) {
    if ((int)threadIdx.x < s) red[threadIdx.x] = fmaxf(red[threadIdx.x], red[threadIdx.x + s]);
    __syncthreads();
  }
  if (threadIdx.x == 0) scales[blockIdx.x] = red[0] / 127.f;
}

// ---------------- stage 2: int8 fake-quant weights -> f16 [F, k*DP] (zero padded) ----------------
__global__ void quantw_kernel(const float* __restrict__ w, const float* __restrict__ scales,
                              int scaleIdx, int k, _Float16* __restrict__ wq) {
  const int Kk = k * DP;
  int idx = blockIdx.x * 256 + threadIdx.x;
  if (idx >= FF * Kk) return;
  int f = idx / Kk;
  int r = idx - f * Kk;
  int j = r / DP;
  int d = r - j * DP;
  float v = 0.f;
  if (d < DD) {
    float s = scales[scaleIdx];
    float x = w[(f * DD + d) * k + j];           // w layout [F, D, k]
    float q = rintf(x / s);
    q = fminf(fmaxf(q, -127.f), 127.f);
    v = q * s;
  }
  wq[idx] = (_Float16)v;
}

// ---------------- stage 3: embedding gather -> f16 [B, SP, DP], padded zeros ----------------
__global__ void embed_kernel(const int* __restrict__ x, const float* __restrict__ emb,
                             _Float16* __restrict__ e) {
  const int pos = blockIdx.x;          // b*SP + s
  const int b = pos / SP;
  const int s = pos - b * SP;
  const int d = threadIdx.x;           // 0..319
  float v = 0.f;
  if (s < SS && d < DD) {
    int tok = x[b * SS + s];
    v = emb[(size_t)tok * DD + d];
  }
  e[(size_t)pos * DP + d] = (_Float16)v;
}

// ---------------- stage 4: conv-as-GEMM (TDM->LDS->WMMA) + quant_relu6 + max-pool ----------------
// Block = (b, fTile of 16), 4 waves. Depth-chunk-outer loop: TDM stages the
// 32-column slab e[b, 0..519, dc*32 .. +32) once (double-buffered); inner
// loops reuse it for all k window offsets j and all 32 t-tiles.
template <int K>
__global__ __launch_bounds__(128)
void conv_wmma_pool_kernel(const _Float16* __restrict__ e,
                           const _Float16* __restrict__ wq,
                           const float* __restrict__ bias,
                           float* __restrict__ h, int kOff) {
  constexpr int Kk  = K * DP;
  constexpr int NDC = DP / 32;                 // 10 depth slabs
  const int b     = blockIdx.x;
  const int fBase = blockIdx.y * 16;
  const int wave  = threadIdx.x >> 5;
  const int lane  = threadIdx.x & 31;
  const int mn    = lane & 15;                 // M row (A) / N col (B,C)
  const int hi    = lane >> 4;                 // lane half -> K/M sub-range

  extern __shared__ char smem[];
  const unsigned int ldsBase = (unsigned int)(uintptr_t)&smem[0];

  const _Float16* eb = e + (size_t)b * SP * DP;
  const unsigned long long ebBytes = (unsigned long long)(uintptr_t)eb;
  const _Float16* wb = wq + (size_t)(fBase + mn) * Kk;

  v8f acc[8];
  #pragma unroll
  for (int i = 0; i < 8; ++i)
    #pragma unroll
    for (int v = 0; v < 8; ++v) acc[i][v] = 0.f;

  // prologue: stage slab 0
  if (wave == 0) {
    tdm_load_2d_f16(ldsBase, ebBytes, DP, SP, 32, SP, DP, 3u, 3u, true);
    __builtin_amdgcn_s_wait_tensorcnt((short)0);
  }
  __syncthreads();

  #pragma unroll 1
  for (int dc = 0; dc < NDC; ++dc) {
    const int cur = dc & 1;
    if ((dc + 1 < NDC) && wave == 0) {
      tdm_load_2d_f16(ldsBase + (unsigned int)((cur ^ 1) * ABUF),
                      ebBytes + (unsigned long long)(dc + 1) * 64ull,
                      DP, SP, 32, SP, DP, 3u, 3u, true);
    }
    const char* A = smem + (size_t)cur * ABUF;   // no pointer table (LDS!)

    #pragma unroll
    for (int j = 0; j < K; ++j) {
      // B fragment from global (L2-hot; 32x16 K-major striping)
      const int kg = j * DP + dc * 32;
      const _Float16* wp = wb + kg + hi * 8;
      v8h blo = *(const v8h*)wp;
      v8h bhi = *(const v8h*)(wp + 16);
      v16h bfrag = __builtin_shufflevector(blo, bhi,
          0,1,2,3,4,5,6,7,8,9,10,11,12,13,14,15);
      __builtin_prefetch((const void*)(wp + 32), 0, 1);   // global_prefetch_b8

      #pragma unroll
      for (int tt = 0; tt < 8; ++tt) {
        const int row = (wave * 8 + tt) * 16 + mn + j;    // t + j
        const char* ap = A + row * ROWB + hi * 16;        // K = 8*hi..
        v8h alo = *(const v8h*)ap;                        // ds_load_b128
        v8h ahi = *(const v8h*)(ap + 32);                 // K = 16+8*hi..
        v16h afrag = __builtin_shufflevector(alo, ahi,
            0,1,2,3,4,5,6,7,8,9,10,11,12,13,14,15);
        acc[tt] = __builtin_amdgcn_wmma_f32_16x16x32_f16(
            false, afrag, false, bfrag, (short)0, acc[tt], false, false);
      }
    }

    if (wave == 0) __builtin_amdgcn_s_wait_tensorcnt((short)0);
    __syncthreads();   // publish next slab / retire reads of current one
  }

  // Raw max over valid t (bias/clip/quant are monotone -> apply after pooling).
  const int tMax = SS - K;
  float wmax = -3.0e38f;
  #pragma unroll
  for (int tt = 0; tt < 8; ++tt) {
    const int tBase = (wave * 8 + tt) * 16;
    #pragma unroll
    for (int v = 0; v < 8; ++v) {
      const int t = tBase + hi * 8 + v;        // C/D layout: M = 8*hi + vgpr
      float val = (t <= tMax) ? acc[tt][v] : -3.0e38f;
      wmax = fmaxf(wmax, val);
    }
  }
  wmax = fmaxf(wmax, __shfl_xor(wmax, 16, 32));   // merge lane halves (same N)

  __shared__ float smax[4][16];
  if (lane < 16) smax[wave][mn] = wmax;
  __syncthreads();
  if (threadIdx.x < 16) {
    float m = fmaxf(fmaxf(smax[0][threadIdx.x], smax[1][threadIdx.x]),
                    fmaxf(smax[2][threadIdx.x], smax[3][threadIdx.x]));
    m += bias[fBase + threadIdx.x];
    m = fminf(fmaxf(m, 0.f), 6.f);               // quant_relu6
    const float qs = 6.f / 255.f;
    m = rintf(m / qs) * qs;
    h[(size_t)b * (3 * FF) + kOff + fBase + threadIdx.x] = m;
  }
}

// ---------------- stage 5: FC (fake-quant weights) + quant_sigmoid ----------------
__global__ void fc_kernel(const float* __restrict__ h, const float* __restrict__ fcw,
                          const float* __restrict__ fcb, const float* __restrict__ scales,
                          float* __restrict__ out) {
  const int b = threadIdx.x;
  if (b >= BB) return;
  const float s = scales[3];
  for (int c = 0; c < CC; ++c) {
    float acc = fcb[c];
    for (int i = 0; i < 3 * FF; ++i) {
      float w = fcw[c * 3 * FF + i];
      float q = fminf(fmaxf(rintf(w / s), -127.f), 127.f) * s;
      acc += h[(size_t)b * (3 * FF) + i] * q;
    }
    float sig = 1.f / (1.f + __expf(-acc));
    out[b * CC + c] = rintf(sig * 255.f) / 255.f;    // quant_sigmoid
  }
}

static inline size_t alignUp(size_t v, size_t a) { return (v + a - 1) & ~(a - 1); }

extern "C" void kernel_launch(void* const* d_in, const int* in_sizes, int n_in,
                              void* d_out, int out_size, void* d_ws, size_t ws_size,
                              hipStream_t stream) {
  // setup_inputs() dict order: x, emb, fcw, fcb, w3, b3, w4, b4, w5, b5
  const int*   x   = (const int*)  d_in[0];
  const float* emb = (const float*)d_in[1];
  const float* fcw = (const float*)d_in[2];
  const float* fcb = (const float*)d_in[3];
  const float* w3  = (const float*)d_in[4];
  const float* b3  = (const float*)d_in[5];
  const float* w4  = (const float*)d_in[6];
  const float* b4  = (const float*)d_in[7];
  const float* w5  = (const float*)d_in[8];
  const float* b5  = (const float*)d_in[9];
  float* out = (float*)d_out;

  // workspace carve-up
  char* ws = (char*)d_ws;
  size_t off = 0;
  float* scales = (float*)(ws + off);             off = alignUp(off + 4 * sizeof(float), 256);
  _Float16* e   = (_Float16*)(ws + off);          off = alignUp(off + (size_t)BB * SP * DP * 2, 256);
  _Float16* wq3 = (_Float16*)(ws + off);          off = alignUp(off + (size_t)FF * 3 * DP * 2, 256);
  _Float16* wq4 = (_Float16*)(ws + off);          off = alignUp(off + (size_t)FF * 4 * DP * 2, 256);
  _Float16* wq5 = (_Float16*)(ws + off);          off = alignUp(off + (size_t)FF * 5 * DP * 2, 256);
  float* h      = (float*)(ws + off);             off = alignUp(off + (size_t)BB * 3 * FF * 4, 256);
  (void)ws_size; (void)in_sizes; (void)n_in; (void)out_size;

  // 1: scales
  scales_kernel<<<4, 256, 0, stream>>>(w3, w4, w5, fcw, scales);

  // 2: weight fake-quant -> f16, K-padded
  quantw_kernel<<<(FF * 3 * DP + 255) / 256, 256, 0, stream>>>(w3, scales, 0, 3, wq3);
  quantw_kernel<<<(FF * 4 * DP + 255) / 256, 256, 0, stream>>>(w4, scales, 1, 4, wq4);
  quantw_kernel<<<(FF * 5 * DP + 255) / 256, 256, 0, stream>>>(w5, scales, 2, 5, wq5);

  // 3: embedding gather -> f16 padded
  embed_kernel<<<BB * SP, DP, 0, stream>>>(x, emb, e);

  // 4: conv + relu6-quant + max-pool (TDM + WMMA), feats ordered k=3,4,5
  dim3 cgrid(BB, FF / 16);
  conv_wmma_pool_kernel<3><<<cgrid, 128, CONV_SMEM, stream>>>(e, wq3, b3, h, 0 * FF);
  conv_wmma_pool_kernel<4><<<cgrid, 128, CONV_SMEM, stream>>>(e, wq4, b4, h, 1 * FF);
  conv_wmma_pool_kernel<5><<<cgrid, 128, CONV_SMEM, stream>>>(e, wq5, b5, h, 2 * FF);

  // 5: FC + quant_sigmoid -> out [64, 2]
  fc_kernel<<<1, BB, 0, stream>>>(h, fcw, fcb, scales, out);
}
